// ProposalLayer_1417339207753
// MI455X (gfx1250) — compile-verified, matrix operand order checked by reading, not run on hardware
//
#include <hip/hip_runtime.h>
#include <hip/hip_bf16.h>
#include <math.h>

namespace {

constexpr int kB     = 8;
constexpr int kFHW   = 50;
constexpr int kA     = 9;
constexpr int kNA    = 22500;          // anchors per image
constexpr int kG     = 20;
constexpr int kPre   = 2000;
constexpr int kPost  = 300;
constexpr int kGX    = (kNA + 255) / 256;        // 88
constexpr int kPBLK  = kB * kGX;                 // 704 partial blocks
constexpr int kRTile = 128;                      // rank rows per block (8 waves * 16)
constexpr int kRGX   = (kNA + kRTile - 1) / kRTile;  // 176
constexpr int kRChunk = 1024;                    // j staged in LDS per outer step

typedef __attribute__((ext_vector_type(16))) __bf16 v16bf;
typedef __attribute__((ext_vector_type(8)))  float  v8f;

// dims = [16*s*sqrt(r), 16*s*sqrt(1/r)] for s in (8,16,32), r in (0.5,1,2); f64->f32 like numpy
__constant__ float c_DW[9] = {
  90.50966799187808f, 128.0f, 181.01933598375618f,
  181.01933598375618f, 256.0f, 362.03867196751236f,
  362.03867196751236f, 512.0f, 724.0773439350247f};
__constant__ float c_DH[9] = {
  181.01933598375618f, 128.0f, 90.50966799187808f,
  362.03867196751236f, 256.0f, 181.01933598375618f,
  724.0773439350247f, 512.0f, 362.03867196751236f};

__device__ __forceinline__ void anchor_corners(int n, float& a0, float& a1,
                                               float& a2, float& a3) {
  int dd = n / 2500;
  int c  = n - dd * 2500;
  float cx = 8.0f + 16.0f * (float)(c / 50);
  float cy = 8.0f + 16.0f * (float)(c % 50);
  float hw = 0.5f * c_DW[dd];
  float hh = 0.5f * c_DH[dd];
  a0 = cx - hw; a1 = cy - hh; a2 = cx + hw; a3 = cy + hh;
}

__device__ __forceinline__ float iou_box(float ax1, float ay1, float ax2, float ay2,
                                         float bx1, float by1, float bx2, float by2) {
  float tlx = fmaxf(ax1, bx1), tly = fmaxf(ay1, by1);
  float brx = fminf(ax2, bx2), bry = fminf(ay2, by2);
  float w = fmaxf(brx - tlx, 0.0f), h = fmaxf(bry - tly, 0.0f);
  float inter = w * h;
  float s1 = (ax2 - ax1) * (ay2 - ay1);
  float s2 = (bx2 - bx1) * (by2 - by1);
  return inter / fmaxf(s1 + s2 - inter, 1e-8f);
}

__device__ __forceinline__ unsigned rotl32(unsigned x, int r) {
  return (x << r) | (x >> (32 - r));
}

// Bit-exact replica of jax.random.uniform(jax.random.key(42), (8,22500)) element [idx].
__device__ __forceinline__ float jax_uniform_180000(unsigned idx) {
  const unsigned H = 90000u;  // half of 8*22500
  unsigned x0, x1;
  if (idx < H) { x0 = idx; x1 = idx + H; } else { x0 = idx - H; x1 = idx; }
  const unsigned ks0 = 0u, ks1 = 42u, ks2 = 0u ^ 42u ^ 0x1BD11BDAu;
  x0 += ks0; x1 += ks1;
#define TF_R(r) { x0 += x1; x1 = rotl32(x1, r); x1 ^= x0; }
  TF_R(13) TF_R(15) TF_R(26) TF_R(6)
  x0 += ks1; x1 += ks2 + 1u;
  TF_R(17) TF_R(29) TF_R(16) TF_R(24)
  x0 += ks2; x1 += ks0 + 2u;
  TF_R(13) TF_R(15) TF_R(26) TF_R(6)
  x0 += ks0; x1 += ks1 + 3u;
  TF_R(17) TF_R(29) TF_R(16) TF_R(24)
  x0 += ks1; x1 += ks2 + 4u;
  TF_R(13) TF_R(15) TF_R(26) TF_R(6)
  x0 += ks2; x1 += ks0 + 5u;
#undef TF_R
  unsigned bits = (idx < H) ? x0 : x1;
  unsigned f = (bits >> 9) | 0x3F800000u;
  return __uint_as_float(f) - 1.0f;
}

// ---------------- kernels ----------------

__global__ void k_zero(int* __restrict__ cnts) {
  if (threadIdx.x < 18) cnts[threadIdx.x] = 0;
}

// scores (channel 9+a) and clipped decoded proposals
__global__ __launch_bounds__(256) void k_prop(const float* __restrict__ pred,
                                              const float* __restrict__ cls,
                                              float* __restrict__ scores,
                                              float* __restrict__ props) {
  int b = blockIdx.y;
  int n = blockIdx.x * 256 + threadIdx.x;
  if (n >= kNA) return;
  int h = n / (kFHW * kA);
  int rem = n - h * (kFHW * kA);
  int w = rem / kA;
  int a = rem - w * kA;
  int sp = h * kFHW + w;
  const int plane = kFHW * kFHW;
  scores[b * kNA + n] = cls[(size_t)(b * 2 * kA + kA + a) * plane + sp];
  float p0 = pred[(size_t)(b * 4 * kA + 4 * a + 0) * plane + sp];
  float p1 = pred[(size_t)(b * 4 * kA + 4 * a + 1) * plane + sp];
  float p2 = pred[(size_t)(b * 4 * kA + 4 * a + 2) * plane + sp];
  float p3 = pred[(size_t)(b * 4 * kA + 4 * a + 3) * plane + sp];
  float a0, a1, a2, a3; anchor_corners(n, a0, a1, a2, a3);
  float acx = (a0 + a2) * 0.5f, acy = (a1 + a3) * 0.5f;
  float aw = a2 - a0, ah = a3 - a1;
  float pcx = acx + p0 * aw;
  float pcy = acy + p1 * ah;
  float pw = aw * expf(p2);
  float ph = ah * expf(p3);
  float x1 = pcx - 0.5f * pw, y1 = pcy - 0.5f * ph;
  float x2 = pcx + 0.5f * pw, y2 = pcy + 0.5f * ph;
  float* o = props + (size_t)(b * kNA + n) * 4;
  o[0] = fminf(fmaxf(x1, 0.0f), 799.0f);
  o[1] = fminf(fmaxf(y1, 0.0f), 799.0f);
  o[2] = fminf(fmaxf(x2, 0.0f), 799.0f);
  o[3] = fminf(fmaxf(y2, 0.0f), 799.0f);
}

// per-anchor max/argmax IoU over 20 GTs -> provisional labels + best-gt index
__global__ __launch_bounds__(256) void k_label(const float* __restrict__ gt,
                                               int* __restrict__ labels,
                                               int* __restrict__ amg) {
  __shared__ float sg[kG * 4];
  int b = blockIdx.y;
  if (threadIdx.x < kG * 4) sg[threadIdx.x] = gt[b * kG * 4 + threadIdx.x];
  __syncthreads();
  int n = blockIdx.x * 256 + threadIdx.x;
  if (n >= kNA) return;
  float a0, a1, a2, a3; anchor_corners(n, a0, a1, a2, a3);
  float best = -1.0f; int bi = 0;
  for (int g = 0; g < kG; g++) {
    float v = iou_box(sg[4 * g], sg[4 * g + 1], sg[4 * g + 2], sg[4 * g + 3],
                      a0, a1, a2, a3);
    if (v > best) { best = v; bi = g; }  // strict > keeps first max (jnp.argmax)
  }
  int lab = -1;
  if (best < 0.3f) lab = 0;
  if (best >= 0.7f) lab = 1;
  labels[b * kNA + n] = lab;
  amg[b * kNA + n] = bi;
}

// per-GT argmax over anchors -> force label 1 (one wave per (b,g))
__global__ __launch_bounds__(32) void k_gtbest(const float* __restrict__ gt,
                                               int* __restrict__ labels) {
  int bg = blockIdx.x;
  int b = bg / kG, g = bg - b * kG;
  int lane = threadIdx.x;
  float g0 = gt[(size_t)(b * kG + g) * 4 + 0];
  float g1 = gt[(size_t)(b * kG + g) * 4 + 1];
  float g2 = gt[(size_t)(b * kG + g) * 4 + 2];
  float g3 = gt[(size_t)(b * kG + g) * 4 + 3];
  float best = -1.0f; int bi = kNA;
  for (int n = lane; n < kNA; n += 32) {
    float a0, a1, a2, a3; anchor_corners(n, a0, a1, a2, a3);
    float v = iou_box(g0, g1, g2, g3, a0, a1, a2, a3);
    if (v > best) { best = v; bi = n; }
  }
  for (int off = 16; off > 0; off >>= 1) {
    float ob = __shfl_down(best, off);
    int oi = __shfl_down(bi, off);
    if (ob > best || (ob == best && oi < bi)) { best = ob; bi = oi; }
  }
  if (lane == 0) labels[b * kNA + bi] = 1;  // benign same-value races across g
}

// inside-anchor override, per-anchor JAX rand value, pos/neg counts
__global__ __launch_bounds__(256) void k_prep(int* __restrict__ labels,
                                              float* __restrict__ rvals,
                                              int* __restrict__ cnts) {
  int b = blockIdx.y;
  int n = blockIdx.x * 256 + threadIdx.x;
  if (n >= kNA) return;
  float a0, a1, a2, a3; anchor_corners(n, a0, a1, a2, a3);
  int lab = labels[b * kNA + n];
  bool inside = (a0 >= 0.0f) && (a1 >= 0.0f) && (a2 <= 800.0f) && (a3 <= 800.0f);
  if (inside) lab = -1;  // reference: labels.at[:, INSIDE_IDX].set(-1)
  labels[b * kNA + n] = lab;
  rvals[b * kNA + n] = jax_uniform_180000((unsigned)(b * kNA + n));
  if (lab == 1) atomicAdd(&cnts[b], 1);
  else if (lab == 0) atomicAdd(&cnts[8 + b], 1);
}

// Stable descending rank of each score via BF16 WMMA row-sum of 0/1 comparison
// tiles: rank_i = #{j : s_j > s_i || (s_j == s_i && j < i)}; rank<2000 -> sel[rank]=i.
__global__ __launch_bounds__(256) void k_topk(const float* __restrict__ scores,
                                              int* __restrict__ sel) {
  __shared__ float sj[kRChunk];
  int b = blockIdx.y;
  int lane = threadIdx.x & 31;
  int wave = threadIdx.x >> 5;
  int rowbase = blockIdx.x * kRTile + wave * 16;
  int m = lane & 15;
  int khalf = lane >> 4;
  int i = rowbase + m;
  float si = (i < kNA) ? scores[b * kNA + i] : 3.0e38f;
  v8f acc = {0.f, 0.f, 0.f, 0.f, 0.f, 0.f, 0.f, 0.f};
  v16bf onesb;
#pragma unroll
  for (int e = 0; e < 16; e++) onesb[e] = (__bf16)1.0f;
  for (int j0 = 0; j0 < kNA; j0 += kRChunk) {
    __syncthreads();
    for (int t = threadIdx.x; t < kRChunk; t += 256) {
      int j = j0 + t;
      sj[t] = (j < kNA) ? scores[b * kNA + j] : -3.0e38f;  // pad never wins
    }
    __syncthreads();
    for (int jc = 0; jc < kRChunk; jc += 32) {
      v16bf av;
#pragma unroll
      for (int e = 0; e < 16; e++) {
        // 16-bit A layout: low lanes own K {0..7,16..23}, high lanes {8..15,24..31}
        int k = ((e & 8) ? (e + 8) : e) + (khalf << 3);
        float sv = sj[jc + k];
        int j = j0 + jc + k;
        bool cmp = (sv > si) || ((sv == si) && (j < i));
        av[e] = (__bf16)(cmp ? 1.0f : 0.0f);
      }
      acc = __builtin_amdgcn_wmma_f32_16x16x32_bf16(false, av, false, onesb,
                                                    (short)0, acc, false, false);
    }
  }
  // D layout: lanes 0-15 hold M=v, lanes 16-31 hold M=v+8; all N columns equal.
  int rr, q;
  if (lane < 8)                      { rr = rowbase + lane;             q = lane; }
  else if (lane >= 16 && lane < 24)  { rr = rowbase + 8 + (lane - 16);  q = lane - 16; }
  else                               { rr = -1;                          q = 0; }
  if (rr >= 0 && rr < kNA) {
    float cv = 0.0f;
#pragma unroll
    for (int e = 0; e < 8; e++) if (q == e) cv = acc[e];
    int rank = (int)(cv + 0.5f);
    if (rank < kPre) sel[b * kPre + rank] = rr;
  }
}

// Same WMMA ranking trick for the random-subsampling double-argsort ranks.
__global__ __launch_bounds__(256) void k_sample(const float* __restrict__ rvals,
                                                const int* __restrict__ labels,
                                                const int* __restrict__ cnts,
                                                int* __restrict__ slab) {
  __shared__ float rj[kRChunk];
  __shared__ int lj[kRChunk];
  int b = blockIdx.y;
  int lane = threadIdx.x & 31;
  int wave = threadIdx.x >> 5;
  int rowbase = blockIdx.x * kRTile + wave * 16;
  int m = lane & 15;
  int khalf = lane >> 4;
  int i = rowbase + m;
  int mylab = (i < kNA) ? labels[b * kNA + i] : -12345;
  float ri = (i < kNA) ? rvals[b * kNA + i] : 0.0f;
  v8f acc = {0.f, 0.f, 0.f, 0.f, 0.f, 0.f, 0.f, 0.f};
  v16bf onesb;
#pragma unroll
  for (int e = 0; e < 16; e++) onesb[e] = (__bf16)1.0f;
  for (int j0 = 0; j0 < kNA; j0 += kRChunk) {
    __syncthreads();
    for (int t = threadIdx.x; t < kRChunk; t += 256) {
      int j = j0 + t;
      rj[t] = (j < kNA) ? rvals[b * kNA + j] : 9.0f;
      lj[t] = (j < kNA) ? labels[b * kNA + j] : -777;
    }
    __syncthreads();
    for (int jc = 0; jc < kRChunk; jc += 32) {
      v16bf av;
#pragma unroll
      for (int e = 0; e < 16; e++) {
        int k = ((e & 8) ? (e + 8) : e) + (khalf << 3);
        float rv = rj[jc + k];
        int lv = lj[jc + k];
        int j = j0 + jc + k;
        bool cmp = (lv == mylab) && ((rv < ri) || ((rv == ri) && (j < i)));
        av[e] = (__bf16)(cmp ? 1.0f : 0.0f);
      }
      acc = __builtin_amdgcn_wmma_f32_16x16x32_bf16(false, av, false, onesb,
                                                    (short)0, acc, false, false);
    }
  }
  int rr, q;
  if (lane < 8)                      { rr = rowbase + lane;             q = lane; }
  else if (lane >= 16 && lane < 24)  { rr = rowbase + 8 + (lane - 16);  q = lane - 16; }
  else                               { rr = -1;                          q = 0; }
  if (rr >= 0 && rr < kNA) {
    float cv = 0.0f;
#pragma unroll
    for (int e = 0; e < 8; e++) if (q == e) cv = acc[e];
    int rank = (int)(cv + 0.5f);
    int lab = labels[b * kNA + rr];
    int np = cnts[b];     if (np > 128) np = 128;        // N_SAMPLES*POS_FRAC
    int lim = 256 - np;
    int nn = cnts[8 + b]; if (nn > lim) nn = lim;
    int o;
    if (lab == 1)      o = (rank < np) ? 1 : -1;
    else if (lab == 0) o = (rank < nn) ? 0 : -1;
    else               o = -1;
    slab[b * kNA + rr] = o;
  }
}

// Greedy NMS over the top-2000 (score-ordered) boxes; one workgroup per image.
__global__ __launch_bounds__(1024) void k_nms(const float* __restrict__ props,
                                              const int* __restrict__ sel,
                                              float* __restrict__ out) {
  __shared__ float bx[kPre * 4];
  __shared__ int supp[kPre];
  int b = blockIdx.x;
  int tid = threadIdx.x;
  for (int t = tid; t < kPre; t += 1024) {
    int idx = sel[b * kPre + t];
    const float* p = props + (size_t)(b * kNA + idx) * 4;
    bx[t * 4 + 0] = p[0]; bx[t * 4 + 1] = p[1];
    bx[t * 4 + 2] = p[2]; bx[t * 4 + 3] = p[3];
    supp[t] = 0;
  }
  for (int t = tid; t < kPost * 4; t += 1024) out[(size_t)b * kPost * 4 + t] = 0.0f;
  __syncthreads();
  for (int i = 0; i < kPre - 1; i++) {
    if (supp[i] == 0) {
      float ax1 = bx[i * 4], ay1 = bx[i * 4 + 1];
      float ax2 = bx[i * 4 + 2], ay2 = bx[i * 4 + 3];
      for (int j = i + 1 + tid; j < kPre; j += 1024) {
        float v = iou_box(ax1, ay1, ax2, ay2,
                          bx[j * 4], bx[j * 4 + 1], bx[j * 4 + 2], bx[j * 4 + 3]);
        if (v > 0.7f) supp[j] = 1;
      }
    }
    __syncthreads();
  }
  // compact unsuppressed in order, floor(), keep first 300
  for (int j = tid; j < kPre; j += 1024) {
    if (supp[j] == 0) {
      int pos = 0;
      for (int mm = 0; mm < j; mm++) pos += (supp[mm] == 0) ? 1 : 0;
      if (pos < kPost) {
        float* o = out + (size_t)b * kPost * 4 + (size_t)pos * 4;
        o[0] = floorf(bx[j * 4 + 0]);
        o[1] = floorf(bx[j * 4 + 1]);
        o[2] = floorf(bx[j * 4 + 2]);
        o[3] = floorf(bx[j * 4 + 3]);
      }
    }
  }
}

// per-anchor loss contributions + fixed-order block partials, int counters
__global__ __launch_bounds__(256) void k_loss(const float* __restrict__ pred,
                                              const float* __restrict__ cls,
                                              const float* __restrict__ gt,
                                              const int* __restrict__ slab,
                                              const int* __restrict__ amg,
                                              int* __restrict__ cnts,
                                              float* __restrict__ clspart,
                                              float* __restrict__ bbpart) {
  __shared__ float sc[256];
  __shared__ float sb[256];
  int b = blockIdx.y;
  int n = blockIdx.x * 256 + threadIdx.x;
  float cv = 0.0f, bv = 0.0f;
  if (n < kNA) {
    int L = slab[b * kNA + n];
    if (L != -1) {
      atomicAdd(&cnts[16], 1);
      if (b == 0) atomicAdd(&cnts[17], 1);
      int h = n / (kFHW * kA);
      int rem = n - h * (kFHW * kA);
      int w = rem / kA;
      int a = rem - w * kA;
      int sp = h * kFHW + w;
      const int plane = kFHW * kFHW;
      float x0 = cls[(size_t)(b * 2 * kA + 2 * a) * plane + sp];
      float x1 = cls[(size_t)(b * 2 * kA + 2 * a + 1) * plane + sp];
      float mx = fmaxf(x0, x1);
      float lse = mx + logf(expf(x0 - mx) + expf(x1 - mx));
      cv = lse - (L == 1 ? x1 : x0);
      if (L == 1) {
        float p0 = pred[(size_t)(b * 4 * kA + 4 * a + 0) * plane + sp];
        float p1 = pred[(size_t)(b * 4 * kA + 4 * a + 1) * plane + sp];
        float p2 = pred[(size_t)(b * 4 * kA + 4 * a + 2) * plane + sp];
        float p3 = pred[(size_t)(b * 4 * kA + 4 * a + 3) * plane + sp];
        float a0, a1, a2, a3; anchor_corners(n, a0, a1, a2, a3);
        float acx = (a0 + a2) * 0.5f, acy = (a1 + a3) * 0.5f;
        float aw = a2 - a0, ah = a3 - a1;
        int g = amg[b * kNA + n];
        const float* gp = gt + (size_t)(b * kG + g) * 4;
        float gcx = (gp[0] + gp[2]) * 0.5f, gcy = (gp[1] + gp[3]) * 0.5f;
        float gw = gp[2] - gp[0], gh = gp[3] - gp[1];
        float dd0 = p0 - (gcx - acx) / aw;
        float dd1 = p1 - (gcy - acy) / ah;
        float dd2 = p2 - logf(gw / aw);
        float dd3 = p3 - logf(gh / ah);
        const float inv = (float)(1.0 / 9.0);
        const float half_inv = (float)(0.5 / 9.0);
        float dv[4] = {dd0, dd1, dd2, dd3};
#pragma unroll
        for (int kk = 0; kk < 4; kk++) {
          float d = dv[kk], ad = fabsf(d);
          bv += (ad < inv) ? 4.5f * d * d : (ad - half_inv);
        }
      }
    }
  }
  sc[threadIdx.x] = cv; sb[threadIdx.x] = bv;
  __syncthreads();
  for (int s = 128; s > 0; s >>= 1) {
    if ((int)threadIdx.x < s) {
      sc[threadIdx.x] += sc[threadIdx.x + s];
      sb[threadIdx.x] += sb[threadIdx.x + s];
    }
    __syncthreads();
  }
  if (threadIdx.x == 0) {
    int pb = blockIdx.y * gridDim.x + blockIdx.x;
    clspart[pb] = sc[0];
    bbpart[pb] = sb[0];
  }
}

__global__ __launch_bounds__(256) void k_final(const float* __restrict__ clspart,
                                               const float* __restrict__ bbpart,
                                               const int* __restrict__ cnts,
                                               float* __restrict__ out) {
  __shared__ float sc[256];
  __shared__ float sb[256];
  float c = 0.0f, bbv = 0.0f;
  for (int i = threadIdx.x; i < kPBLK; i += 256) { c += clspart[i]; bbv += bbpart[i]; }
  sc[threadIdx.x] = c; sb[threadIdx.x] = bbv;
  __syncthreads();
  for (int s = 128; s > 0; s >>= 1) {
    if ((int)threadIdx.x < s) {
      sc[threadIdx.x] += sc[threadIdx.x + s];
      sb[threadIdx.x] += sb[threadIdx.x + s];
    }
    __syncthreads();
  }
  if (threadIdx.x == 0) {
    int mc = cnts[16]; if (mc < 1) mc = 1;   // mask.sum()
    int rc = cnts[17]; if (rc < 1) rc = 1;   // (labels[0] >= 0).sum()
    out[kB * kPost * 4 + 0] = sb[0] / (float)rc;  // bbox_loss
    out[kB * kPost * 4 + 1] = sc[0] / (float)mc;  // cls_loss
  }
}

}  // namespace

extern "C" void kernel_launch(void* const* d_in, const int* in_sizes, int n_in,
                              void* d_out, int out_size, void* d_ws, size_t ws_size,
                              hipStream_t stream) {
  (void)in_sizes; (void)n_in; (void)out_size; (void)ws_size;
  const float* pred = (const float*)d_in[0];
  const float* cls  = (const float*)d_in[1];
  const float* gt   = (const float*)d_in[2];
  float* out = (float*)d_out;

  // workspace layout (~6.6 MB)
  float* fws    = (float*)d_ws;
  float* scores = fws;                                   // B*NA
  float* props  = scores + (size_t)kB * kNA;             // B*NA*4
  float* rvals  = props + (size_t)kB * kNA * 4;          // B*NA
  float* clspart = rvals + (size_t)kB * kNA;             // PBLK
  float* bbpart  = clspart + kPBLK;                      // PBLK
  int* sel    = (int*)(bbpart + kPBLK);                  // B*PRE
  int* labels = sel + kB * kPre;                         // B*NA
  int* amg    = labels + kB * kNA;                       // B*NA
  int* slab   = amg + kB * kNA;                          // B*NA
  int* cnts   = slab + kB * kNA;                         // 18 ints

  k_zero<<<1, 32, 0, stream>>>(cnts);
  k_prop<<<dim3(kGX, kB), 256, 0, stream>>>(pred, cls, scores, props);
  k_label<<<dim3(kGX, kB), 256, 0, stream>>>(gt, labels, amg);
  k_gtbest<<<kB * kG, 32, 0, stream>>>(gt, labels);
  k_prep<<<dim3(kGX, kB), 256, 0, stream>>>(labels, rvals, cnts);
  k_topk<<<dim3(kRGX, kB), 256, 0, stream>>>(scores, sel);
  k_sample<<<dim3(kRGX, kB), 256, 0, stream>>>(rvals, labels, cnts, slab);
  k_nms<<<kB, 1024, 0, stream>>>(props, sel, out);
  k_loss<<<dim3(kGX, kB), 256, 0, stream>>>(pred, cls, gt, slab, amg, cnts,
                                            clspart, bbpart);
  k_final<<<1, 256, 0, stream>>>(clspart, bbpart, cnts, out);
}
